// MarlinQuantLinear_77910706750105
// MI455X (gfx1250) — matrix operand order repaired and checked by reading, not compiled, not run on hardware
//
#include <hip/hip_runtime.h>

// ---------------------------------------------------------------------------
// GPTQ int4 (sym, zp=8, group=128) x fp32 -> fp32, M=8192 K=4096 N=4096.
// Compute-bound (AI ~1000 flop/byte) => bf16 WMMA with f32 accumulation.
//
// 3-kernel plan (ws >= 96MiB):
//   1) cvt_x_bf16:    x f32 -> bf16 [M][K]      (done once, not per N-block)
//   2) dequant_wt:    qweight int4 -> bf16 W^T [N][K]  (done once, not per M-block)
//   3) gemm_bf16:     pure-copy staging via global_load_async_to_lds_b128
//                     (ASYNCcnt + s_wait_asynccnt), WMMA inner loop, ~no VALU.
//                     K-loop forced to unroll 1 + offset-toggled double buffer
//                     so accumulators stay pinned (no cross-unroll v_mov_b64).
// Both bf16 operands (64MiB + 32MiB) fit in the 192MB L2 together, so the
// cross-block re-reads in the GEMM are L2 hits. HBM floor ~= 16us; bf16 matrix
// compute ~= 55us dominates -> compute-bound.
// Fallback (small ws): round-1 fused dequant-in-LDS kernel.
// ---------------------------------------------------------------------------

typedef __attribute__((ext_vector_type(16))) __bf16 v16bf;
typedef __attribute__((ext_vector_type(8)))  float  v8f;
typedef int v4i_vec __attribute__((__vector_size__(16)));   // matches builtin param

#define M_DIM 8192
#define K_DIM 4096
#define N_DIM 4096
#define BLK_MN 128
#define BLK_K  64
#define KTILES (K_DIM / BLK_K)   // 64
#define THREADS 256
#define BUF_ELEMS (BLK_MN * BLK_K)   // 8192 elements = 16 KB

#define GAS __attribute__((address_space(1)))
#define LAS __attribute__((address_space(3)))

#if __has_builtin(__builtin_amdgcn_global_load_async_to_lds_b128)
#define USE_ASYNC_LDS 1
#else
#define USE_ASYNC_LDS 0
#endif

__device__ __forceinline__ void wait_async0() {
#if __has_builtin(__builtin_amdgcn_s_wait_asynccnt)
  __builtin_amdgcn_s_wait_asynccnt(0);
#else
  asm volatile("s_wait_asynccnt 0x0" ::: "memory");
#endif
}

// round-to-nearest-even f32 -> bf16, packed pair into one u32
__device__ __forceinline__ unsigned int pack_bf16x2(float lo, float hi) {
  unsigned int ulo = __float_as_uint(lo);
  unsigned int uhi = __float_as_uint(hi);
  ulo = (ulo + 0x7FFFu + ((ulo >> 16) & 1u)) >> 16;
  uhi = (uhi + 0x7FFFu + ((uhi >> 16) & 1u)) & 0xFFFF0000u;
  return ulo | uhi;
}

// dequantize nibble pair (2p, 2p+1) of q: (nib - 8) * s, packed to 2x bf16
__device__ __forceinline__ unsigned int dq2(unsigned int q, int p, float s) {
  int nlo = (int)((q >> (8 * p))     & 0xFu) - 8;
  int nhi = (int)((q >> (8 * p + 4)) & 0xFu) - 8;
  return pack_bf16x2((float)nlo * s, (float)nhi * s);
}

union AB16 { v16bf v; uint4 u[2]; };

// ---------------------------------------------------------------------------
// Kernel 1: x f32 -> bf16, [M][K] row-major. 8 elems/thread, fully coalesced.
// ---------------------------------------------------------------------------
__global__ __launch_bounds__(THREADS, 1)
void cvt_x_bf16(const float* __restrict__ x, unsigned short* __restrict__ xb) {
  const size_t base = ((size_t)blockIdx.x * THREADS + threadIdx.x) * 8;
  float4 v0 = *(const float4*)(x + base);
  float4 v1 = *(const float4*)(x + base + 4);
  uint4 u = make_uint4(pack_bf16x2(v0.x, v0.y), pack_bf16x2(v0.z, v0.w),
                       pack_bf16x2(v1.x, v1.y), pack_bf16x2(v1.z, v1.w));
  *(uint4*)(xb + base) = u;
}

// ---------------------------------------------------------------------------
// Kernel 2: qweight [K/8][N] int32 -> W^T bf16 [N][K]. One int32 (8 k-nibbles
// of one column n) per thread; reads coalesced over n, 16B write per thread.
// ---------------------------------------------------------------------------
__global__ __launch_bounds__(THREADS, 1)
void dequant_wt_bf16(const int* __restrict__ qweight,
                     const float* __restrict__ scales,
                     unsigned short* __restrict__ wt) {
  const int gid = blockIdx.x * THREADS + threadIdx.x;  // [0, (K/8)*N)
  const int n  = gid & (N_DIM - 1);
  const int kp = gid >> 12;                            // N = 2^12
  const unsigned int q = (unsigned int)qweight[gid];   // qweight[kp][n]
  const float s = scales[(size_t)(kp >> 4) * N_DIM + n];  // group = kp*8/128
  uint4 u = make_uint4(dq2(q, 0, s), dq2(q, 1, s), dq2(q, 2, s), dq2(q, 3, s));
  *(uint4*)(wt + (size_t)n * K_DIM + kp * 8) = u;
}

// ---------------------------------------------------------------------------
// Kernel 3: bf16 GEMM. Block 128x128, K-tile 64, 8 waves (4x2), 32x64 C/wave.
// Staging is a pure 16KB+16KB copy per K-tile: async global->LDS (b128).
// ---------------------------------------------------------------------------
__global__ __launch_bounds__(THREADS, 1)
void gemm_bf16_wmma(const unsigned short* __restrict__ Abf,   // [M][K]
                    const unsigned short* __restrict__ Btf,   // [N][K]
                    float* __restrict__ out) {
  __shared__ unsigned short As[2 * BUF_ELEMS];   // 2 x 16 KB
  __shared__ unsigned short Bs[2 * BUF_ELEMS];   // 2 x 16 KB

  const int tid = threadIdx.x;
  const int n0  = blockIdx.x * BLK_MN;
  const int m0  = blockIdx.y * BLK_MN;

  // copy mapping: 1024 16B-chunks per operand tile; thread t does chunk
  // t + 256*i, i=0..3 -> row = t/8 + 32*i, k-offset = (t%8)*8 elements.
  const int crow = tid >> 3;        // 0..31
  const int ck   = (tid & 7) * 8;   // element offset in K

  const unsigned short* agp = Abf + (size_t)(m0 + crow) * K_DIM + ck;
  const unsigned short* bgp = Btf + (size_t)(n0 + crow) * K_DIM + ck;

  auto issue_tile = [&](int kt, int woff) {
    const int koff = kt * BLK_K;
    unsigned short* al = &As[woff + crow * BLK_K + ck];
    unsigned short* bl = &Bs[woff + crow * BLK_K + ck];
#if USE_ASYNC_LDS
    #pragma unroll
    for (int i = 0; i < 4; ++i) {
      __builtin_amdgcn_global_load_async_to_lds_b128(
          (GAS v4i_vec*)(unsigned short*)(agp + koff + (size_t)(32 * i) * K_DIM),
          (LAS v4i_vec*)(al + i * 32 * BLK_K), 0, 0);
      __builtin_amdgcn_global_load_async_to_lds_b128(
          (GAS v4i_vec*)(unsigned short*)(bgp + koff + (size_t)(32 * i) * K_DIM),
          (LAS v4i_vec*)(bl + i * 32 * BLK_K), 0, 0);
    }
#else
    uint4 ra[4], rb[4];
    #pragma unroll
    for (int i = 0; i < 4; ++i) {
      ra[i] = *(const uint4*)(agp + koff + (size_t)(32 * i) * K_DIM);
      rb[i] = *(const uint4*)(bgp + koff + (size_t)(32 * i) * K_DIM);
    }
    #pragma unroll
    for (int i = 0; i < 4; ++i) {
      *(uint4*)(al + i * 32 * BLK_K) = ra[i];
      *(uint4*)(bl + i * 32 * BLK_K) = rb[i];
    }
#endif
  };

  // ---- wave / output mapping ----
  const int wave  = tid >> 5;
  const int lane  = tid & 31;
  const int wm    = wave & 3;     // 4 wave-rows of 32 M
  const int wn    = wave >> 2;    // 2 wave-cols of 64 N
  const int lhalf = lane >> 4;
  const int l16   = lane & 15;

  // per-lane LDS element offsets of the frag runs (without buffer offset)
  const int a_base = (wm * 32 + l16) * BLK_K + lhalf * 8;
  const int b_base = (wn * 64 + l16) * BLK_K + lhalf * 8;

  v8f acc[2][4];
  #pragma unroll
  for (int mt = 0; mt < 2; ++mt)
    #pragma unroll
    for (int nt = 0; nt < 4; ++nt)
      acc[mt][nt] = (v8f)(0.0f);

  issue_tile(0, 0);
  wait_async0();
  __syncthreads();

  int roff = 0;
  int woff = BUF_ELEMS;
  #pragma unroll 1
  for (int kt = 0; kt < KTILES; ++kt) {
    if (kt + 1 < KTILES) issue_tile(kt + 1, woff);   // overlaps with WMMA

    #pragma unroll
    for (int ks = 0; ks < 2; ++ks) {
      AB16 afrag[2], bfrag[4];
      // 16-bit A 16x32 layout: lanes 0-15 K={0..7,16..23}, lanes 16-31
      // K={8..15,24..31}: two contiguous 8-elem runs per lane.
      #pragma unroll
      for (int mt = 0; mt < 2; ++mt) {
        const unsigned short* p = &As[roff + a_base + mt * 16 * BLK_K + ks * 32];
        afrag[mt].u[0] = *(const uint4*)(p);
        afrag[mt].u[1] = *(const uint4*)(p + 16);
      }
      // 16-bit B 32x16 layout: lane = column, lanes 0-15 K={0..7,16..23},
      // lanes 16-31 K={8..15,24..31}: two contiguous 8-elem runs per lane.
      #pragma unroll
      for (int nt = 0; nt < 4; ++nt) {
        const unsigned short* p = &Bs[roff + b_base + nt * 16 * BLK_K + ks * 32];
        bfrag[nt].u[0] = *(const uint4*)(p);
        bfrag[nt].u[1] = *(const uint4*)(p + 16);
      }
      #pragma unroll
      for (int mt = 0; mt < 2; ++mt)
        #pragma unroll
        for (int nt = 0; nt < 4; ++nt)
          acc[mt][nt] = __builtin_amdgcn_wmma_f32_16x16x32_bf16(
              false, afrag[mt].v, false, bfrag[nt].v,
              (short)0, acc[mt][nt], false, false);
    }

    // toggle double buffer (single loop body -> accumulators stay pinned)
    const int t = roff; roff = woff; woff = t;
    if (kt + 1 < KTILES) wait_async0();
    __syncthreads();
  }

  // epilogue: C layout lane=N (l16), VGPR v = M row (+8 for upper lane half)
  float* obase = out + (size_t)(m0 + wm * 32 + lhalf * 8) * N_DIM
                     + (n0 + wn * 64 + l16);
  #pragma unroll
  for (int mt = 0; mt < 2; ++mt) {
    #pragma unroll
    for (int nt = 0; nt < 4; ++nt) {
      float* p = obase + (size_t)(mt * 16) * N_DIM + nt * 16;
      #pragma unroll
      for (int v = 0; v < 8; ++v)
        p[(size_t)v * N_DIM] = acc[mt][nt][v];
    }
  }
}

// ---------------------------------------------------------------------------
// Fallback: fused dequant-in-LDS kernel (round-1 version) if ws is too small.
// ---------------------------------------------------------------------------
__global__ __launch_bounds__(THREADS, 1)
void gptq_gemm_fused(const float* __restrict__ x,
                     const int*   __restrict__ qweight,
                     const float* __restrict__ scales,
                     float*       __restrict__ out) {
  __shared__ unsigned short As[2 * BUF_ELEMS];
  __shared__ unsigned short Bs[2 * BUF_ELEMS];

  const int tid = threadIdx.x;
  const int n0  = blockIdx.x * BLK_MN;
  const int m0  = blockIdx.y * BLK_MN;

  const int a_row = tid >> 1;
  const int a_col = (tid & 1) * 32;
  const int b_row = tid >> 5;
  const int b_col = (tid & 31) * 4;

  const float* a_gptr = x + (size_t)(m0 + a_row) * K_DIM + a_col;
  const int*   b_gptr = qweight + (size_t)b_row * N_DIM + (n0 + b_col);
  const float* s_gptr = scales + (n0 + b_col);

  unsigned int a_regs[16];
  int4   b_regs;
  float4 s_regs;

  auto load_tile = [&](int kt) {
    const float* ap = a_gptr + kt * BLK_K;
    #pragma unroll
    for (int j = 0; j < 8; ++j) {
      float4 v = *(const float4*)(ap + j * 4);
      a_regs[2 * j + 0] = pack_bf16x2(v.x, v.y);
      a_regs[2 * j + 1] = pack_bf16x2(v.z, v.w);
    }
    b_regs = *(const int4*)(b_gptr + (size_t)kt * 8 * N_DIM);
    s_regs = *(const float4*)(s_gptr + (size_t)(kt >> 1) * N_DIM);
  };

  auto store_tile = [&](int woff) {
    unsigned short* adst = &As[woff + a_row * BLK_K + a_col];
    #pragma unroll
    for (int j = 0; j < 4; ++j) {
      *(uint4*)(adst + j * 8) = make_uint4(a_regs[4 * j], a_regs[4 * j + 1],
                                           a_regs[4 * j + 2], a_regs[4 * j + 3]);
    }
    const unsigned int qv[4] = {(unsigned int)b_regs.x, (unsigned int)b_regs.y,
                                (unsigned int)b_regs.z, (unsigned int)b_regs.w};
    const float sv[4] = {s_regs.x, s_regs.y, s_regs.z, s_regs.w};
    #pragma unroll
    for (int c = 0; c < 4; ++c) {
      const unsigned int q = qv[c];
      const float s = sv[c];
      *(uint4*)(&Bs[woff + (b_col + c) * BLK_K + b_row * 8]) =
          make_uint4(dq2(q, 0, s), dq2(q, 1, s), dq2(q, 2, s), dq2(q, 3, s));
    }
  };

  const int wave = tid >> 5, lane = tid & 31;
  const int wm = wave & 3, wn = wave >> 2;
  const int lhalf = lane >> 4, l16 = lane & 15;
  const int a_base = (wm * 32 + l16) * BLK_K + lhalf * 8;
  const int b_base = (wn * 64 + l16) * BLK_K + lhalf * 8;

  v8f acc[2][4];
  #pragma unroll
  for (int mt = 0; mt < 2; ++mt)
    #pragma unroll
    for (int nt = 0; nt < 4; ++nt)
      acc[mt][nt] = (v8f)(0.0f);

  load_tile(0);
  store_tile(0);
  __syncthreads();

  int roff = 0, woff = BUF_ELEMS;
  #pragma unroll 1
  for (int kt = 0; kt < KTILES; ++kt) {
    if (kt + 1 < KTILES) load_tile(kt + 1);
    #pragma unroll
    for (int ks = 0; ks < 2; ++ks) {
      AB16 afrag[2], bfrag[4];
      #pragma unroll
      for (int mt = 0; mt < 2; ++mt) {
        const unsigned short* p = &As[roff + a_base + mt * 16 * BLK_K + ks * 32];
        afrag[mt].u[0] = *(const uint4*)(p);
        afrag[mt].u[1] = *(const uint4*)(p + 16);
      }
      #pragma unroll
      for (int nt = 0; nt < 4; ++nt) {
        const unsigned short* p = &Bs[roff + b_base + nt * 16 * BLK_K + ks * 32];
        bfrag[nt].u[0] = *(const uint4*)(p);
        bfrag[nt].u[1] = *(const uint4*)(p + 16);
      }
      #pragma unroll
      for (int mt = 0; mt < 2; ++mt)
        #pragma unroll
        for (int nt = 0; nt < 4; ++nt)
          acc[mt][nt] = __builtin_amdgcn_wmma_f32_16x16x32_bf16(
              false, afrag[mt].v, false, bfrag[nt].v,
              (short)0, acc[mt][nt], false, false);
    }
    if (kt + 1 < KTILES) store_tile(woff);
    const int t = roff; roff = woff; woff = t;
    __syncthreads();
  }

  float* obase = out + (size_t)(m0 + wm * 32 + lhalf * 8) * N_DIM
                     + (n0 + wn * 64 + l16);
  #pragma unroll
  for (int mt = 0; mt < 2; ++mt)
    #pragma unroll
    for (int nt = 0; nt < 4; ++nt) {
      float* p = obase + (size_t)(mt * 16) * N_DIM + nt * 16;
      #pragma unroll
      for (int v = 0; v < 8; ++v)
        p[(size_t)v * N_DIM] = acc[mt][nt][v];
    }
}

extern "C" void kernel_launch(void* const* d_in, const int* in_sizes, int n_in,
                              void* d_out, int out_size, void* d_ws, size_t ws_size,
                              hipStream_t stream) {
  const float* x  = (const float*)d_in[0];
  const int*   qw = (const int*)d_in[1];
  const float* sc = (const float*)d_in[2];
  float* out = (float*)d_out;

  const size_t needA = (size_t)M_DIM * K_DIM * sizeof(unsigned short);  // 64 MiB
  const size_t needB = (size_t)N_DIM * K_DIM * sizeof(unsigned short);  // 32 MiB
  dim3 grid(N_DIM / BLK_MN, M_DIM / BLK_MN);   // (32, 64)

  if (ws_size >= needA + needB) {
    unsigned short* Abf = (unsigned short*)d_ws;
    unsigned short* Btf = Abf + (size_t)M_DIM * K_DIM;
    cvt_x_bf16<<<(size_t)M_DIM * K_DIM / (THREADS * 8), THREADS, 0, stream>>>(x, Abf);
    dequant_wt_bf16<<<(K_DIM / 8) * N_DIM / THREADS, THREADS, 0, stream>>>(qw, sc, Btf);
    gemm_bf16_wmma<<<grid, dim3(THREADS), 0, stream>>>(Abf, Btf, out);
  } else {
    gptq_gemm_fused<<<grid, dim3(THREADS), 0, stream>>>(x, qw, sc, out);
  }
}
